// ReferenceSelectiveScan1D_84911503442582
// MI455X (gfx1250) — compile-verified, matrix operand order checked by reading, not run on hardware
//
#include <hip/hip_runtime.h>
#include <hip/hip_bf16.h>
#include <math.h>

#define DIMX   1024
#define INNER  512
#define DSTATE 8
#define DTRANK 64
#define BATCH  4
#define SEQ    4096
#define ROWS   (BATCH * SEQ)          // 16384
#define NPROJ  (DTRANK + 2 * DSTATE)  // 80
#define NCHUNK 64
#define CHUNK  (SEQ / NCHUNK)         // 64
#define KTILES (DIMX / 32)            // 32  (K=1024 for both big GEMMs)

typedef __attribute__((ext_vector_type(2)))  float v2f;
typedef __attribute__((ext_vector_type(8)))  float v8f;
typedef __bf16 bf16_t;
typedef __attribute__((ext_vector_type(8)))  bf16_t v8bf;
typedef __attribute__((ext_vector_type(16))) bf16_t v16bf;

__device__ __forceinline__ float softplusf(float v) {
    return v > 20.f ? v : log1pf(expf(v));
}
__device__ __forceinline__ float clip01(float v) {
    return fminf(fmaxf(v, 1e-4f), 1.0f);
}
__device__ __forceinline__ float siluf(float v) {
    return v / (1.f + expf(-v));
}

// ===========================================================================
// fp32 -> (bf16 hi, bf16 lo) split, written in WMMA fragment order.
// Fragment f = ((tile16 * KTILES + kt) * 32 + lane), 16 bf16 elements/lane:
//   elem e<8  -> K = kt*32 + (lane>=16)*8 + e
//   elem e>=8 -> K = kt*32 + 16 + (lane>=16)*8 + (e-8)
// One thread handles one 8-element half-fragment chunk (16B store).
// ===========================================================================
__global__ void cvt_split_frag(const float* __restrict__ X,
                               bf16_t* __restrict__ hi, bf16_t* __restrict__ lo) {
    const int tid  = blockIdx.x * blockDim.x + threadIdx.x;  // R*1024/8 chunks
    const int half = tid & 1;
    const int frag = tid >> 1;
    const int lane = frag & 31;
    const int kt   = (frag >> 5) & (KTILES - 1);
    const int t16  = frag >> 10;                 // row tile (rows/16)
    const int row  = t16 * 16 + (lane & 15);
    const int kb   = kt * 32 + half * 16 + (lane >> 4) * 8;

    const float* src = X + (size_t)row * DIMX + kb;
    v8bf h, l;
#pragma unroll
    for (int i = 0; i < 8; i++) {
        float xv = src[i];
        bf16_t hb = (bf16_t)xv;
        h[i] = hb;
        l[i] = (bf16_t)(xv - (float)hb);
    }
    *((v8bf*)hi + tid) = h;
    *((v8bf*)lo + tid) = l;
}

__device__ __forceinline__ v16bf load_frag(const bf16_t* __restrict__ base, size_t fragIdx) {
    const v8bf* p = (const v8bf*)base + fragIdx * 2;
    v8bf a = p[0];
    v8bf b = p[1];
    return __builtin_shufflevector(a, b, 0, 1, 2, 3, 4, 5, 6, 7,
                                   8, 9, 10, 11, 12, 13, 14, 15);
}

// ===========================================================================
// Split-bf16 GEMM: C[m,n] = sum_k A[m,k]*W[n,k], K = 1024.
// A,W pre-split hi/lo bf16, fragment-ordered. 3-term compensated product:
//   acc += Ahi*Bhi + Alo*Bhi + Ahi*Blo      (~fp32 accuracy)
// One wave -> 32(M) x 64(N): 2x4 tiles of 16x16, fp32 accumulators.
// (2x4 keeps VGPR usage ~130 -> no scratch spills, room to pipeline loads.)
// ===========================================================================
__global__ void gemm_bf16_split(const bf16_t* __restrict__ Ahi, const bf16_t* __restrict__ Alo,
                                const bf16_t* __restrict__ Whi, const bf16_t* __restrict__ Wlo,
                                float* __restrict__ C, int ldc) {
    const int wave = threadIdx.x >> 5;
    const int lane = threadIdx.x & 31;
    const int mt0  = blockIdx.x * 2;                                   // 2 M-tiles
    const int nt0  = (blockIdx.y * (blockDim.x >> 5) + wave) * 4;      // 4 N-tiles

    v8f acc[2][4] = {};
    for (int kt = 0; kt < KTILES; kt++) {
        v16bf ah[2], al[2];
#pragma unroll
        for (int m = 0; m < 2; m++) {
            size_t f = ((size_t)(mt0 + m) * KTILES + kt) * 32 + lane;
            ah[m] = load_frag(Ahi, f);
            al[m] = load_frag(Alo, f);
        }
#pragma unroll
        for (int n = 0; n < 4; n++) {
            size_t f = ((size_t)(nt0 + n) * KTILES + kt) * 32 + lane;
            v16bf bh = load_frag(Whi, f);
            v16bf bl = load_frag(Wlo, f);
#pragma unroll
            for (int m = 0; m < 2; m++) {
                acc[m][n] = __builtin_amdgcn_wmma_f32_16x16x32_bf16(false, ah[m], false, bh, (short)0, acc[m][n], false, false);
                acc[m][n] = __builtin_amdgcn_wmma_f32_16x16x32_bf16(false, al[m], false, bh, (short)0, acc[m][n], false, false);
                acc[m][n] = __builtin_amdgcn_wmma_f32_16x16x32_bf16(false, ah[m], false, bl, (short)0, acc[m][n], false, false);
            }
        }
    }

    const int rowOff = (lane >> 4) * 8;   // lanes 16-31 -> M+8
    const int col    = lane & 15;
#pragma unroll
    for (int m = 0; m < 2; m++)
#pragma unroll
        for (int n = 0; n < 4; n++) {
            float* cp = C + (size_t)(mt0 * 16 + m * 16 + rowOff) * ldc + nt0 * 16 + n * 16 + col;
#pragma unroll
            for (int r = 0; r < 8; r++)
                cp[(size_t)r * ldc] = acc[m][n][r];
        }
}

// ===========================================================================
// Exact-precision f32 WMMA (16x16x4) for the small GEMMs.
// EPI==0: plain store. EPI==1: C = clip(softplus(acc + bias[n]), 1e-4, 1)
// ===========================================================================
template <int EPI>
__global__ void gemm_wmma_16x64(const float* __restrict__ A, int lda,
                                const float* __restrict__ W, int ldw,
                                float* __restrict__ C, int ldc,
                                int K, const float* __restrict__ bias) {
    const int wave = threadIdx.x >> 5;
    const int lane = threadIdx.x & 31;
    const int mBase = blockIdx.x * 16;
    const int nBase = (blockIdx.y * (blockDim.x >> 5) + wave) * 64;

    const int rowA = mBase + (lane & 15);
    const int nb   = nBase + (lane & 15);
    const int koff = (lane >> 4) * 2;

    const float* ap  = A + rowA * lda + koff;
    const float* wp0 = W + (nb +  0) * ldw + koff;
    const float* wp1 = W + (nb + 16) * ldw + koff;
    const float* wp2 = W + (nb + 32) * ldw + koff;
    const float* wp3 = W + (nb + 48) * ldw + koff;

    v8f acc0 = {}, acc1 = {}, acc2 = {}, acc3 = {};
    for (int k0 = 0; k0 < K; k0 += 4) {
        v2f a  = *(const v2f*)(ap  + k0);
        v2f b0 = *(const v2f*)(wp0 + k0);
        v2f b1 = *(const v2f*)(wp1 + k0);
        v2f b2 = *(const v2f*)(wp2 + k0);
        v2f b3 = *(const v2f*)(wp3 + k0);
        acc0 = __builtin_amdgcn_wmma_f32_16x16x4_f32(false, a, false, b0, (short)0, acc0, false, false);
        acc1 = __builtin_amdgcn_wmma_f32_16x16x4_f32(false, a, false, b1, (short)0, acc1, false, false);
        acc2 = __builtin_amdgcn_wmma_f32_16x16x4_f32(false, a, false, b2, (short)0, acc2, false, false);
        acc3 = __builtin_amdgcn_wmma_f32_16x16x4_f32(false, a, false, b3, (short)0, acc3, false, false);
    }

    const int rowC = mBase + ((lane >> 4) << 3);
    const int col  = lane & 15;
    float bb0 = 0.f, bb1 = 0.f, bb2 = 0.f, bb3 = 0.f;
    if (EPI == 1) {
        bb0 = bias[nBase + col +  0];
        bb1 = bias[nBase + col + 16];
        bb2 = bias[nBase + col + 32];
        bb3 = bias[nBase + col + 48];
    }
#pragma unroll
    for (int r = 0; r < 8; r++) {
        float* cp = C + (size_t)(rowC + r) * ldc + nBase + col;
        if (EPI == 0) {
            cp[0]  = acc0[r];
            cp[16] = acc1[r];
            cp[32] = acc2[r];
            cp[48] = acc3[r];
        } else {
            cp[0]  = clip01(softplusf(acc0[r] + bb0));
            cp[16] = clip01(softplusf(acc1[r] + bb1));
            cp[32] = clip01(softplusf(acc2[r] + bb2));
            cp[48] = clip01(softplusf(acc3[r] + bb3));
        }
    }
}

// 16x16-per-wave variant for narrow N (proj, N = 80). 4 waves = 4 M-tiles.
__global__ void gemm_wmma_16x16(const float* __restrict__ A, int lda,
                                const float* __restrict__ W, int ldw,
                                float* __restrict__ C, int ldc, int K) {
    const int wave = threadIdx.x >> 5;
    const int lane = threadIdx.x & 31;
    const int mBase = (blockIdx.x * 4 + wave) * 16;
    const int nBase = blockIdx.y * 16;

    const int rowA = mBase + (lane & 15);
    const int nb   = nBase + (lane & 15);
    const int koff = (lane >> 4) * 2;

    const float* ap = A + rowA * lda + koff;
    const float* wp = W + nb * ldw + koff;

    v8f acc = {};
    for (int k0 = 0; k0 < K; k0 += 4) {
        v2f a = *(const v2f*)(ap + k0);
        v2f b = *(const v2f*)(wp + k0);
        acc = __builtin_amdgcn_wmma_f32_16x16x4_f32(false, a, false, b, (short)0, acc, false, false);
    }
    const int rowC = mBase + ((lane >> 4) << 3);
    const int col  = lane & 15;
#pragma unroll
    for (int r = 0; r < 8; r++)
        C[(size_t)(rowC + r) * ldc + nBase + col] = acc[r];
}

// ===========================================================================
// Depthwise conv (K=3, pad 1, cross-correlation) along L + SiLU.
// ===========================================================================
__global__ void dwconv_silu(const float* __restrict__ h,
                            const float* __restrict__ Kx,
                            const float* __restrict__ Kz,
                            float* __restrict__ xb,
                            float* __restrict__ ycat) {
    const int tid = blockIdx.x * blockDim.x + threadIdx.x;   // ROWS*DIMX
    const int c1024 = tid & (DIMX - 1);
    const int row   = tid >> 10;
    const int l     = row & (SEQ - 1);
    const bool isx  = c1024 < INNER;
    const int c     = isx ? c1024 : c1024 - INNER;
    const float* kern = (isx ? Kx : Kz) + c * 3;

    const size_t base = (size_t)row * DIMX + c1024;
    float um = (l > 0)       ? h[base - DIMX] : 0.f;
    float u0 = h[base];
    float up = (l < SEQ - 1) ? h[base + DIMX] : 0.f;
    float v  = um * kern[0] + u0 * kern[1] + up * kern[2];
    float s  = siluf(v);
    if (isx) xb[(size_t)row * INNER + c] = s;
    else     ycat[(size_t)row * DIMX + INNER + c] = s;
}

__global__ void bc_tanh(const float* __restrict__ proj,
                        float* __restrict__ bterm, float* __restrict__ cterm) {
    const int tid = blockIdx.x * blockDim.x + threadIdx.x;   // ROWS*16
    const int j   = tid & 15;
    const int row = tid >> 4;
    float v = tanhf(proj[(size_t)row * NPROJ + DTRANK + j]);
    if (j < DSTATE) bterm[(size_t)row * DSTATE + j] = v;
    else            cterm[(size_t)row * DSTATE + (j - DSTATE)] = v;
}

// ===========================================================================
// Chunked selective scan (64 chunks of 64): A-phase per-chunk transfer,
// B-phase serial combine over chunks, C-phase replay + output.
// ===========================================================================
__global__ void scan_chunk(const float* __restrict__ delta,
                           const float* __restrict__ xb,
                           const float* __restrict__ bterm,
                           const float* __restrict__ A_log,
                           float* __restrict__ cA, float* __restrict__ cB) {
    const int tid = blockIdx.x * blockDim.x + threadIdx.x;   // B*INNER*NCHUNK
    const int c  = tid % INNER;
    const int ch = (tid / INNER) % NCHUNK;
    const int b  = tid / (INNER * NCHUNK);

    float a[DSTATE], st[DSTATE], ap[DSTATE];
#pragma unroll
    for (int s = 0; s < DSTATE; s++) {
        a[s]  = softplusf(A_log[c * DSTATE + s]) + 1e-4f;
        st[s] = 0.f;
        ap[s] = 1.f;
    }
    const int l0 = ch * CHUNK;
    for (int i = 0; i < CHUNK; i++) {
        const int row = b * SEQ + l0 + i;
        const float d  = delta[(size_t)row * INNER + c];
        const float xv = xb[(size_t)row * INNER + c];
#pragma unroll
        for (int s = 0; s < DSTATE; s++) {
            float dec = clip01(expf(-d * a[s]));
            float drv = (1.f - dec) * bterm[(size_t)row * DSTATE + s] * xv;
            st[s] = dec * st[s] + drv;
            ap[s] *= dec;
        }
    }
    const size_t off = (((size_t)b * INNER + c) * NCHUNK + ch) * DSTATE;
#pragma unroll
    for (int s = 0; s < DSTATE; s++) { cA[off + s] = ap[s]; cB[off + s] = st[s]; }
}

__global__ void scan_prefix(const float* __restrict__ cA,
                            const float* __restrict__ cB,
                            float* __restrict__ cInit) {
    const int tid = blockIdx.x * blockDim.x + threadIdx.x;   // B*INNER*DSTATE
    const int s = tid % DSTATE;
    const int c = (tid / DSTATE) % INNER;
    const int b = tid / (DSTATE * INNER);
    const size_t base = ((size_t)b * INNER + c) * NCHUNK * DSTATE + s;
    float carry = 0.f;
    for (int k = 0; k < NCHUNK; k++) {
        cInit[base + (size_t)k * DSTATE] = carry;
        carry = cA[base + (size_t)k * DSTATE] * carry + cB[base + (size_t)k * DSTATE];
    }
}

__global__ void scan_apply(const float* __restrict__ delta,
                           const float* __restrict__ xb,
                           const float* __restrict__ bterm,
                           const float* __restrict__ cterm,
                           const float* __restrict__ A_log,
                           const float* __restrict__ Dvec,
                           const float* __restrict__ cInit,
                           float* __restrict__ ycat) {
    const int tid = blockIdx.x * blockDim.x + threadIdx.x;   // B*INNER*NCHUNK
    const int c  = tid % INNER;
    const int ch = (tid / INNER) % NCHUNK;
    const int b  = tid / (INNER * NCHUNK);

    float a[DSTATE], st[DSTATE];
    const size_t off = (((size_t)b * INNER + c) * NCHUNK + ch) * DSTATE;
#pragma unroll
    for (int s = 0; s < DSTATE; s++) {
        a[s]  = softplusf(A_log[c * DSTATE + s]) + 1e-4f;
        st[s] = cInit[off + s];
    }
    const float dd = Dvec[c];
    const int l0 = ch * CHUNK;
    for (int i = 0; i < CHUNK; i++) {
        const int row = b * SEQ + l0 + i;
        const float d  = delta[(size_t)row * INNER + c];
        const float xv = xb[(size_t)row * INNER + c];
        float y = dd * xv;
#pragma unroll
        for (int s = 0; s < DSTATE; s++) {
            float dec = clip01(expf(-d * a[s]));
            float drv = (1.f - dec) * bterm[(size_t)row * DSTATE + s] * xv;
            st[s] = dec * st[s] + drv;
            y += st[s] * cterm[(size_t)row * DSTATE + s];
        }
        ycat[(size_t)row * DIMX + c] = y;
    }
}

// ===========================================================================
extern "C" void kernel_launch(void* const* d_in, const int* in_sizes, int n_in,
                              void* d_out, int out_size, void* d_ws, size_t ws_size,
                              hipStream_t stream) {
    const float* x      = (const float*)d_in[0];
    const float* W_in   = (const float*)d_in[1];
    const float* Kx     = (const float*)d_in[2];
    const float* Kz     = (const float*)d_in[3];
    const float* W_xprj = (const float*)d_in[4];
    const float* W_dt   = (const float*)d_in[5];
    const float* b_dt   = (const float*)d_in[6];
    const float* A_log  = (const float*)d_in[7];
    const float* Dvec   = (const float*)d_in[8];
    const float* W_out  = (const float*)d_in[9];
    float* out = (float*)d_out;

    // fp32 workspace
    float* wf    = (float*)d_ws;
    float* h     = wf;                          // ROWS*1024
    float* xb    = h     + (size_t)ROWS * DIMX;
    float* ycat  = xb    + (size_t)ROWS * INNER;
    float* proj  = ycat  + (size_t)ROWS * DIMX;
    float* delta = proj  + (size_t)ROWS * NPROJ;
    float* bterm = delta + (size_t)ROWS * INNER;
    float* cterm = bterm + (size_t)ROWS * DSTATE;
    float* cA    = cterm + (size_t)ROWS * DSTATE;
    float* cB    = cA    + (size_t)BATCH * INNER * NCHUNK * DSTATE;
    float* cIn   = cB    + (size_t)BATCH * INNER * NCHUNK * DSTATE;

    // bf16 split workspace (fragment-ordered)
    bf16_t* xhi = (bf16_t*)(cIn + (size_t)BATCH * INNER * NCHUNK * DSTATE);
    bf16_t* xlo = xhi + (size_t)ROWS * DIMX;
    bf16_t* wih = xlo + (size_t)ROWS * DIMX;
    bf16_t* wil = wih + (size_t)DIMX * DIMX;
    bf16_t* woh = wil + (size_t)DIMX * DIMX;
    bf16_t* wol = woh + (size_t)DIMX * DIMX;
    // h (64MB fp32) is dead after dwconv; alias for ycat hi/lo (2 x 32MB bf16)
    bf16_t* ych = (bf16_t*)h;
    bf16_t* ycl = ych + (size_t)ROWS * DIMX;

    // 0) split inputs into bf16 hi/lo fragments
    cvt_split_frag<<<(ROWS * DIMX / 8) / 256, 256, 0, stream>>>(x, xhi, xlo);
    cvt_split_frag<<<(DIMX * DIMX / 8) / 256, 256, 0, stream>>>(W_in, wih, wil);
    cvt_split_frag<<<(DIMX * DIMX / 8) / 256, 256, 0, stream>>>(W_out, woh, wol);

    // 1) h = x @ W_in^T  (split-bf16 WMMA, ~fp32 accuracy)
    gemm_bf16_split<<<dim3(ROWS / 32, DIMX / 256), 128, 0, stream>>>(
        xhi, xlo, wih, wil, h, DIMX);

    // 2) depthwise conv + SiLU -> xb, ycat[:,512:]
    dwconv_silu<<<(ROWS * DIMX) / 256, 256, 0, stream>>>(h, Kx, Kz, xb, ycat);

    // 3) proj = xb @ W_xproj^T   (16384x512 * 512x80, f32 WMMA)
    gemm_wmma_16x16<<<dim3(ROWS / 64, NPROJ / 16), 128, 0, stream>>>(
        xb, INNER, W_xprj, INNER, proj, NPROJ, INNER);

    // 4) b_term / c_term = tanh(proj[:, 64:80])
    bc_tanh<<<(ROWS * 16) / 256, 256, 0, stream>>>(proj, bterm, cterm);

    // 5) delta = clip(softplus(dt_raw @ W_dt^T + b_dt))  (K=64, N=512, f32 WMMA)
    gemm_wmma_16x64<1><<<dim3(ROWS / 16, INNER / 256), 128, 0, stream>>>(
        proj, NPROJ, W_dt, DTRANK, delta, INNER, DTRANK, b_dt);

    // 6-8) chunked scan
    scan_chunk<<<(BATCH * INNER * NCHUNK) / 256, 256, 0, stream>>>(
        delta, xb, bterm, A_log, cA, cB);
    scan_prefix<<<(BATCH * INNER * DSTATE) / 256, 256, 0, stream>>>(cA, cB, cIn);
    scan_apply<<<(BATCH * INNER * NCHUNK) / 256, 256, 0, stream>>>(
        delta, xb, bterm, cterm, A_log, Dvec, cIn, ycat);

    // 9) split [y, zb] and run out GEMM on the bf16 pipe
    cvt_split_frag<<<(ROWS * DIMX / 8) / 256, 256, 0, stream>>>(ycat, ych, ycl);
    gemm_bf16_split<<<dim3(ROWS / 32, DIMX / 256), 128, 0, stream>>>(
        ych, ycl, woh, wol, out, DIMX);
}
